// MultiHeadSelfAttention_30047591202923
// MI455X (gfx1250) — matrix operand order repaired and checked by reading, not compiled, not run on hardware
//
#include <hip/hip_runtime.h>
#include <hip/hip_bf16.h>

// ---------------------------------------------------------------------------
// Problem constants (from reference): B=4, T=2048, C=1024, H=16, Dk=64
// ---------------------------------------------------------------------------
#define BB     4
#define TT     2048
#define CC     1024
#define HH     16
#define DK     64
#define MM     (BB * TT)          // 8192 rows of flattened [B*T, C]
#define QTILES (TT / 16)          // 128 query tiles per (b,h)

typedef __attribute__((ext_vector_type(16))) __bf16         v16bf;
typedef __attribute__((ext_vector_type(16))) unsigned short v16u;
typedef __attribute__((ext_vector_type(8)))  float          v8f;

// fp32 -> bf16 bits, round-to-nearest-even
__device__ __forceinline__ unsigned short f2bf(float f) {
    unsigned u = __builtin_bit_cast(unsigned, f);
    if ((u & 0x7fffffffu) > 0x7f800000u) return (unsigned short)0x7fc0u; // NaN
    unsigned r = u + 0x7fffu + ((u >> 16) & 1u);
    return (unsigned short)(r >> 16);
}

__device__ __forceinline__ v8f wmma_bf16(v16u a, v16u b, v8f c) {
    union { v16u u; v16bf h; } A, Bv;
    A.u = a; Bv.u = b;
    return __builtin_amdgcn_wmma_f32_16x16x32_bf16(
        false, A.h, false, Bv.h, (short)0, c, false, false);
}

// A fragment (16x32 bf16, MxK): lane l holds row M=l%16.
// elem e -> K=k0+16*(l/16)+e
__device__ __forceinline__ v16u load_afrag(const unsigned short* rowp, int k0, int hi) {
    v16u a;
    const unsigned short* p = rowp + k0 + 16 * hi;
#pragma unroll
    for (int e = 0; e < 16; ++e) a[e] = p[e];
    return a;
}

// B fragment (32x16 bf16, KxN): lane l holds col N=l%16, elem e -> K=k0+16*(l/16)+e
__device__ __forceinline__ v16u load_bfrag(const unsigned short* colp, int k0, int hi) {
    v16u b;
    const unsigned short* p = colp + k0 + 16 * hi;
#pragma unroll
    for (int e = 0; e < 16; ++e) b[e] = p[e];
    return b;
}

// ---------------------------------------------------------------------------
// Kernel 1: fp32 -> bf16 cast
// ---------------------------------------------------------------------------
__global__ void k_cast(const float* __restrict__ src, unsigned short* __restrict__ dst, int n) {
    int i = blockIdx.x * blockDim.x + threadIdx.x;
    if (i < n) dst[i] = f2bf(src[i]);
}

// ---------------------------------------------------------------------------
// Kernel 2: QKV projection, 16x64 output per wave (A-fragment reused 4x).
//   grid(1024,1,3) x 256 thr -> 8192 waves/z, tile = 512 Mtiles x 16 Ntiles(64)
//   Q,K head-major bf16 [B,H,T,DK]; V transposed bf16 [B,H,DK,T].
// ---------------------------------------------------------------------------
__global__ void __launch_bounds__(256)
k_qkv(const unsigned short* __restrict__ xb,
      const unsigned short* __restrict__ wqb, const unsigned short* __restrict__ wkb,
      const unsigned short* __restrict__ wvb,
      const float* __restrict__ bq, const float* __restrict__ bk, const float* __restrict__ bv,
      unsigned short* __restrict__ qdst, unsigned short* __restrict__ kdst,
      unsigned short* __restrict__ vdst) {
    const int lane = threadIdx.x & 31, wid = threadIdx.x >> 5;
    const int hi = lane >> 4, ln = lane & 15;
    const int z = blockIdx.z;

    const int tile = blockIdx.x * 8 + wid;         // [0, 8192)
    const int tm   = tile >> 4;                    // M tile (512)
    const int tn64 = tile & 15;                    // N block of 64 == head index

    const unsigned short* W   = (z == 0) ? wqb : (z == 1) ? wkb : wvb;
    const float*          bi  = (z == 0) ? bq  : (z == 1) ? bk  : bv;
    unsigned short*       dst = (z == 0) ? qdst : (z == 1) ? kdst : vdst;

    const unsigned short* xrow = xb + (size_t)(tm * 16 + ln) * CC;
    const unsigned short* wrow[4];
#pragma unroll
    for (int s = 0; s < 4; ++s)
        wrow[s] = W + (size_t)(tn64 * 64 + s * 16 + ln) * CC;

    v8f acc[4] = {};
#pragma unroll 2
    for (int k0 = 0; k0 < CC; k0 += 32) {
        __builtin_prefetch(xrow + k0 + 128, 0, 1);
        const v16u a = load_afrag(xrow, k0, hi);
#pragma unroll
        for (int s = 0; s < 4; ++s)
            acc[s] = wmma_bf16(a, load_bfrag(wrow[s], k0, hi), acc[s]);
    }

#pragma unroll
    for (int s = 0; s < 4; ++s) {
        const int j = tn64 * 64 + s * 16 + ln;     // output column
        const float bz = bi[j];
        const int h = tn64, d = s * 16 + ln;
#pragma unroll
        for (int r = 0; r < 8; ++r) {
            const int i = tm * 16 + r + 8 * hi;    // global row
            const int b = i >> 11, t = i & (TT - 1);
            const unsigned short hv = f2bf(acc[s][r] + bz);
            if (z == 2) dst[(((size_t)(b * HH + h) * DK + d) * TT) + t] = hv;
            else        dst[(((size_t)(b * HH + h) * TT + t) * DK) + d] = hv;
        }
    }
}

// ---------------------------------------------------------------------------
// Kernel 3: scores = QK^T/8 (causal) -> raw scores + online row max/sum.
// Block = 8 waves sharing one bh, 8 consecutive query tiles. K tiles are
// staged through LDS with async global->LDS copies (double-buffered,
// overlapped with compute), cutting K traffic 8x. Q fragments hoisted.
// grid 1024 x 256 thr.
// ---------------------------------------------------------------------------
__global__ void __launch_bounds__(256)
k_scores(const unsigned short* __restrict__ qb_, const unsigned short* __restrict__ kb_,
         float* __restrict__ attn, float* __restrict__ statm, float* __restrict__ stats) {
    __shared__ unsigned short kbuf[2][16][72];     // 16 keys x 64 dims, padded rows

    const int lane = threadIdx.x & 31, wid = threadIdx.x >> 5;
    const int hi = lane >> 4, ln = lane & 15;

    const int bh     = blockIdx.x >> 4;            // b*H + h
    const int qgroup = blockIdx.x & 15;
    const int qt     = qgroup * 8 + wid;           // this wave's query tile
    const int ktmax  = qgroup * 8 + 7;             // block-uniform key-tile bound

    const unsigned short* Qrow = qb_ + ((size_t)bh * TT + qt * 16 + ln) * DK;
    const char* kglob = (const char*)(kb_ + (size_t)bh * TT * DK);
    float* arow = attn + (size_t)bh * TT * TT;

    // Loop-invariant Q fragments (Dk = 64 -> two K=32 fragments)
    const v16u qf0 = load_afrag(Qrow, 0, hi);
    const v16u qf1 = load_afrag(Qrow, 32, hi);

    // wave 0 stages one 16x64 bf16 K tile (2 KiB) via 4 async b128 copies
    auto issue_tile = [&](int kt, int slot) {
        const unsigned ldsbase = (unsigned)(uintptr_t)&kbuf[slot][0][0];
#pragma unroll
        for (int c = 0; c < 4; ++c) {
            const int ec  = c * 256 + lane * 8;            // element idx in tile
            const int key = ec >> 6, dim = ec & 63;
            const unsigned lds = ldsbase + (unsigned)(key * 72 + dim) * 2;
            const unsigned long long g =
                (unsigned long long)(uintptr_t)(kglob + ((size_t)kt * 1024 + ec) * 2);
            asm volatile("global_load_async_to_lds_b128 %0, %1, off"
                         :: "v"(lds), "v"(g) : "memory");
        }
    };

    float mrun[8], srun[8];
#pragma unroll
    for (int r = 0; r < 8; ++r) { mrun[r] = -__builtin_inff(); srun[r] = 0.f; }

    if (wid == 0) issue_tile(0, 0);
    for (int kt = 0; kt <= ktmax; ++kt) {
        if (wid == 0) asm volatile("s_wait_asynccnt 0x0" ::: "memory");
        __syncthreads();                            // tile kt visible to all
        if (wid == 0 && kt < ktmax) issue_tile(kt + 1, (kt + 1) & 1);

        if (kt <= qt) {                             // wave-uniform; EXEC stays full
            v16u b0, b1;
            const unsigned short* p0 = &kbuf[kt & 1][ln][16 * hi];
            const unsigned short* p1 = &kbuf[kt & 1][ln][32 + 16 * hi];
#pragma unroll
            for (int e = 0; e < 16; ++e) { b0[e] = p0[e]; b1[e] = p1[e]; }
            v8f c = {};
            c = wmma_bf16(qf0, b0, c);
            c = wmma_bf16(qf1, b1, c);

#pragma unroll
            for (int r = 0; r < 8; ++r) {
                float s = c[r] * 0.125f;            // 1/sqrt(64)
                if (kt == qt && ln > (r + 8 * hi)) s = -__builtin_inff();
                float tmax = s;
                tmax = fmaxf(tmax, __shfl_xor(tmax, 1));
                tmax = fmaxf(tmax, __shfl_xor(tmax, 2));
                tmax = fmaxf(tmax, __shfl_xor(tmax, 4));
                tmax = fmaxf(tmax, __shfl_xor(tmax, 8));
                const float mnew = fmaxf(mrun[r], tmax);
                float e = __expf(s - mnew);
                e += __shfl_xor(e, 1);
                e += __shfl_xor(e, 2);
                e += __shfl_xor(e, 4);
                e += __shfl_xor(e, 8);
                srun[r] = srun[r] * __expf(mrun[r] - mnew) + e;
                mrun[r] = mnew;
                arow[(size_t)(qt * 16 + r + 8 * hi) * TT + kt * 16 + ln] = s;
            }
        }
    }

    if (ln == 0) {
#pragma unroll
        for (int r = 0; r < 8; ++r) {
            const int q = qt * 16 + r + 8 * hi;
            statm[(size_t)bh * TT + q] = mrun[r];
            stats[(size_t)bh * TT + q] = srun[r];
        }
    }
}

// ---------------------------------------------------------------------------
// Kernel 4: normalize in place: attn = exp(raw - m)/sum; zero above diagonal.
// ---------------------------------------------------------------------------
__global__ void k_norm(float* __restrict__ attn,
                       const float* __restrict__ statm, const float* __restrict__ stats) {
    const size_t i  = (size_t)blockIdx.x * blockDim.x + threadIdx.x;
    const size_t kk = i & (TT - 1);
    const size_t q  = (i >> 11) & (TT - 1);
    const size_t bh = i >> 22;
    float o = 0.f;
    if (kk <= q) {
        const float m = statm[bh * TT + q];
        const float s = stats[bh * TT + q];
        o = __expf(attn[i] - m) / s;
    }
    attn[i] = o;
}

// ---------------------------------------------------------------------------
// Kernel 5: head_out = attn @ V.  One wave per (bh, query-tile), computing the
// FULL 16x64 head slice: the f32 attn A-fragment is read once and reused for
// 4 WMMAs (was 4x redundant reads of the 1 GiB attn tensor).
// grid 1024 x 256 thr = 8192 waves.
// ---------------------------------------------------------------------------
__global__ void __launch_bounds__(256)
k_av(const float* __restrict__ attn, const unsigned short* __restrict__ vtb,
     unsigned short* __restrict__ hout) {
    const int lane = threadIdx.x & 31, wid = threadIdx.x >> 5;
    const int hi = lane >> 4, ln = lane & 15;

    const int idx = blockIdx.x * 8 + wid;
    const int qt  = idx & (QTILES - 1);
    const int bh  = idx >> 7;

    const float* arow = attn + (size_t)bh * TT * TT + (size_t)(qt * 16 + ln) * TT;
    const unsigned short* vcol[4];
#pragma unroll
    for (int s = 0; s < 4; ++s)
        vcol[s] = vtb + ((size_t)bh * DK + s * 16 + ln) * TT;

    v8f acc[4] = {};
    const int nk = (qt + 2) >> 1;                  // ceil((qt+1)/2) chunks of 32 keys
    for (int kc = 0; kc < nk; ++kc) {
        const int k0 = kc * 32;
        v16u a;
        const float* pa = arow + k0 + 16 * hi;
#pragma unroll
        for (int e = 0; e < 16; ++e) a[e] = f2bf(pa[e]);
#pragma unroll
        for (int s = 0; s < 4; ++s)
            acc[s] = wmma_bf16(a, load_bfrag(vcol[s], k0, hi), acc[s]);
    }

    const int b = bh >> 4, h = bh & (HH - 1);
#pragma unroll
    for (int s = 0; s < 4; ++s) {
        const int col = h * DK + s * 16 + ln;
#pragma unroll
        for (int r = 0; r < 8; ++r) {
            const int t = qt * 16 + r + 8 * hi;
            hout[((size_t)(b * TT + t) * CC) + col] = f2bf(acc[s][r]);
        }
    }
}

// ---------------------------------------------------------------------------
// Kernel 6: out = head_out @ Wo^T + bo, 16x64 per wave, f32 to d_out.
// ---------------------------------------------------------------------------
__global__ void __launch_bounds__(256)
k_oproj(const unsigned short* __restrict__ hb, const unsigned short* __restrict__ wob,
        const float* __restrict__ bo, float* __restrict__ outp) {
    const int lane = threadIdx.x & 31, wid = threadIdx.x >> 5;
    const int hi = lane >> 4, ln = lane & 15;

    const int tile = blockIdx.x * 8 + wid;
    const int tm   = tile >> 4;
    const int tn64 = tile & 15;

    const unsigned short* arow = hb + (size_t)(tm * 16 + ln) * CC;
    const unsigned short* wrow[4];
#pragma unroll
    for (int s = 0; s < 4; ++s)
        wrow[s] = wob + (size_t)(tn64 * 64 + s * 16 + ln) * CC;

    v8f acc[4] = {};
#pragma unroll 2
    for (int k0 = 0; k0 < CC; k0 += 32) {
        __builtin_prefetch(arow + k0 + 128, 0, 1);
        const v16u a = load_afrag(arow, k0, hi);
#pragma unroll
        for (int s = 0; s < 4; ++s)
            acc[s] = wmma_bf16(a, load_bfrag(wrow[s], k0, hi), acc[s]);
    }

#pragma unroll
    for (int s = 0; s < 4; ++s) {
        const int j = tn64 * 64 + s * 16 + ln;
        const float bz = bo[j];
#pragma unroll
        for (int r = 0; r < 8; ++r) {
            const int i = tm * 16 + r + 8 * hi;
            outp[(size_t)i * CC + j] = acc[s][r] + bz;
        }
    }
}

// ---------------------------------------------------------------------------
// Launcher. Workspace (~89 MiB): xb, 4x W bf16, Qb, Kb, Vtb, stats, hout.
// ---------------------------------------------------------------------------
extern "C" void kernel_launch(void* const* d_in, const int* in_sizes, int n_in,
                              void* d_out, int out_size, void* d_ws, size_t ws_size,
                              hipStream_t stream) {
    const float* x  = (const float*)d_in[0];
    // d_in[1] = attn_mask (causal, deterministic) -- unused
    const float* Wq = (const float*)d_in[2];
    const float* bq = (const float*)d_in[3];
    const float* Wk = (const float*)d_in[4];
    const float* bk = (const float*)d_in[5];
    const float* Wv = (const float*)d_in[6];
    const float* bv = (const float*)d_in[7];
    const float* Wo = (const float*)d_in[8];
    const float* bo = (const float*)d_in[9];

    float* outp = (float*)d_out;
    float* attn = outp + (size_t)MM * CC;          // attn region of d_out

    char* ws = (char*)d_ws;
    size_t off = 0;
    unsigned short* xb   = (unsigned short*)(ws + off); off += (size_t)MM * CC * 2;
    unsigned short* wqb  = (unsigned short*)(ws + off); off += (size_t)CC * CC * 2;
    unsigned short* wkb  = (unsigned short*)(ws + off); off += (size_t)CC * CC * 2;
    unsigned short* wvb  = (unsigned short*)(ws + off); off += (size_t)CC * CC * 2;
    unsigned short* wob  = (unsigned short*)(ws + off); off += (size_t)CC * CC * 2;
    unsigned short* Qb   = (unsigned short*)(ws + off); off += (size_t)MM * CC * 2;
    unsigned short* Kb   = (unsigned short*)(ws + off); off += (size_t)MM * CC * 2;
    unsigned short* Vtb  = (unsigned short*)(ws + off); off += (size_t)MM * CC * 2;
    float*          statm = (float*)(ws + off);     off += (size_t)BB * HH * TT * 4;
    float*          stats = (float*)(ws + off);     off += (size_t)BB * HH * TT * 4;
    unsigned short* hout = (unsigned short*)(ws + off);

    // 1) casts
    k_cast<<<(MM * CC + 255) / 256, 256, 0, stream>>>(x, xb, MM * CC);
    k_cast<<<(CC * CC + 255) / 256, 256, 0, stream>>>(Wq, wqb, CC * CC);
    k_cast<<<(CC * CC + 255) / 256, 256, 0, stream>>>(Wk, wkb, CC * CC);
    k_cast<<<(CC * CC + 255) / 256, 256, 0, stream>>>(Wv, wvb, CC * CC);
    k_cast<<<(CC * CC + 255) / 256, 256, 0, stream>>>(Wo, wob, CC * CC);

    // 2) Q/K/V projections: 8192 waves per matrix (16x64 tiles)
    k_qkv<<<dim3(1024, 1, 3), 256, 0, stream>>>(xb, wqb, wkb, wvb, bq, bk, bv, Qb, Kb, Vtb);

    // 3) causal scores + online stats (async K staging): 8192 waves
    k_scores<<<1024, 256, 0, stream>>>(Qb, Kb, attn, statm, stats);

    // 4) softmax normalize (elementwise over 268M)
    k_norm<<<(unsigned)((size_t)BB * HH * TT * TT / 256), 256, 0, stream>>>(attn, statm, stats);

    // 5) attn @ V: 8192 waves (16x64 per wave, attn read once)
    k_av<<<1024, 256, 0, stream>>>(attn, Vtb, hout);

    // 6) output projection: 8192 waves
    k_oproj<<<1024, 256, 0, stream>>>(hout, wob, bo, outp);
}
